// MoE_Layer_1554778161484
// MI455X (gfx1250) — matrix operand order repaired and checked by reading, not compiled, not run on hardware
//
#include <hip/hip_runtime.h>
#include <math.h>

// ---------------------------------------------------------------------------
// MoE layer for MI455X (gfx1250, wave32, WMMA).
// Routed (top-2) expert compute with v_wmma_f32_16x16x32_bf16:
//   - 4x less math than the reference's dense-all-experts path
//   - bf16 A/B with f32 accumulation (chip's high-throughput matrix path)
//   - 32-row x 1024-col blocks (16 waves): halves per-block B-matrix L2
//     traffic vs 16-row tiles; waves w and w+8 share B addresses (L0 reuse)
// Weights are pre-transposed to column-major bf16 so every WMMA fragment is
// plain contiguous 16B per-lane global loads in one shared k order, k = 16*(lane/16) + s, for both A and B.
// ---------------------------------------------------------------------------

#define NTOK   8192
#define DIM    1024
#define NEXP   8
#define GHID   32
#define NPAIR  (NTOK * 2)
#define MROWS  32              // rows per GEMM block
#define GWAVES 16              // waves per GEMM block (512 threads)

typedef __attribute__((ext_vector_type(16))) __bf16 v16bf;
typedef __attribute__((ext_vector_type(8)))  float  v8f;

union FragBF {            // 32 bytes = 16 bf16 = one WMMA A or B fragment/lane
    v16bf v;
    uint4 q[2];
};

__device__ __forceinline__ unsigned short f32_to_bf16(float f) {
    unsigned int u = __float_as_uint(f);
    unsigned int r = u + 0x7FFFu + ((u >> 16) & 1u);   // round-to-nearest-even
    return (unsigned short)(r >> 16);
}

// ------------------------------- utility -----------------------------------
__global__ void zero_f32(float* p, int n) {
    int i = blockIdx.x * blockDim.x + threadIdx.x;
    if (i < n) p[i] = 0.0f;
}
__global__ void zero_i32(int* p, int n) {
    int i = blockIdx.x * blockDim.x + threadIdx.x;
    if (i < n) p[i] = 0;
}
__global__ void conv_x_bf16(const float* __restrict__ x, unsigned short* __restrict__ xb, int n) {
    int i = blockIdx.x * blockDim.x + threadIdx.x;
    if (i < n) xb[i] = f32_to_bf16(x[i]);
}

// Transpose-convert W[e][r][c] (f32, row-major) -> WT[e][c][r] (bf16)
__global__ void transpose_w_bf16(const float* __restrict__ W, unsigned short* __restrict__ WT) {
    __shared__ float tile[32][33];
    const int e  = blockIdx.z;
    const int r0 = blockIdx.y * 32;
    const int c0 = blockIdx.x * 32;
    const float*    Wp = W  + ((size_t)e << 20);
    unsigned short* Tp = WT + ((size_t)e << 20);
    const int tx = threadIdx.x, ty = threadIdx.y;       // block (32,8)
    #pragma unroll
    for (int i = 0; i < 32; i += 8)
        tile[ty + i][tx] = Wp[(size_t)(r0 + ty + i) * DIM + c0 + tx];
    __syncthreads();
    #pragma unroll
    for (int i = 0; i < 32; i += 8)
        Tp[(size_t)(c0 + ty + i) * DIM + r0 + tx] = f32_to_bf16(tile[tx][ty + i]);
}

// ------------------------------- gating ------------------------------------
// g[n,h] = tanh(x[n,:] @ gw1[:,h] + gb1[h]);  lanes of a wave share n.
__global__ void gate_hidden(const float* __restrict__ x, const float* __restrict__ gw1,
                            const float* __restrict__ gb1, float* __restrict__ g) {
    int idx = blockIdx.x * blockDim.x + threadIdx.x;    // n*32 + h
    int h = idx & (GHID - 1);
    int n = idx >> 5;
    const float* xr = x + (size_t)n * DIM;
    float s = gb1[h];
    #pragma unroll 4
    for (int d = 0; d < DIM; ++d) s = fmaf(xr[d], gw1[d * GHID + h], s);
    g[idx] = tanhf(s);
}

// Per token: logits -> softmax -> top2 -> normalized weights + routing lists.
__global__ void gate_route(const float* __restrict__ g, const float* __restrict__ gw2,
                           const float* __restrict__ gb2, float* __restrict__ wpair,
                           int* __restrict__ list, int* __restrict__ cnt) {
    int n = blockIdx.x * blockDim.x + threadIdx.x;
    if (n >= NTOK) return;
    const float* gr = g + (size_t)n * GHID;
    float logits[NEXP];
    #pragma unroll
    for (int j = 0; j < NEXP; ++j) {
        float s = gb2[j];
        #pragma unroll
        for (int h = 0; h < GHID; ++h) s = fmaf(gr[h], gw2[h * NEXP + j], s);
        logits[j] = s;
    }
    float m = logits[0];
    #pragma unroll
    for (int j = 1; j < NEXP; ++j) m = fmaxf(m, logits[j]);
    float p[NEXP], sum = 0.f;
    #pragma unroll
    for (int j = 0; j < NEXP; ++j) { p[j] = __expf(logits[j] - m); sum += p[j]; }
    float inv = 1.f / sum;
    #pragma unroll
    for (int j = 0; j < NEXP; ++j) p[j] *= inv;
    // top-2 (first-max tie rule like lax.top_k)
    int i0 = 0;
    #pragma unroll
    for (int j = 1; j < NEXP; ++j) if (p[j] > p[i0]) i0 = j;
    int i1 = (i0 == 0) ? 1 : 0;
    #pragma unroll
    for (int j = 0; j < NEXP; ++j) if (j != i0 && p[j] > p[i1]) i1 = j;
    float denom = p[i0] + p[i1] + 1e-6f;
    float w0 = p[i0] / denom, w1 = p[i1] / denom;
    wpair[n * 2 + 0] = w0;
    wpair[n * 2 + 1] = w1;
    int pos0 = atomicAdd(&cnt[i0], 1); list[i0 * NTOK + pos0] = n * 2 + 0;
    int pos1 = atomicAdd(&cnt[i1], 1); list[i1 * NTOK + pos1] = n * 2 + 1;
}

// --------------------------- expert GEMM 1 + LN ----------------------------
// Block: 16 waves, expert blockIdx.y, rows [m0, m0+32), cols 0..1023.
// Wave w: row strip (w>>3)*16, col group (w&7)*128 (8 WMMA tiles).
// h = LN(x@W1 + b1) -> ReLU -> bf16 hbuf[pair].
__global__ __launch_bounds__(512)
void expert_gemm1_ln(const unsigned short* __restrict__ xb,    // [NTOK][DIM] bf16
                     const unsigned short* __restrict__ w1t,   // [E][out][in] bf16
                     const float* __restrict__ b1,             // [E][DIM]
                     const float* __restrict__ ln_g, const float* __restrict__ ln_b,
                     const int* __restrict__ cnt, const int* __restrict__ list,
                     unsigned short* __restrict__ hbuf) {      // [NPAIR][DIM] bf16
    const int e  = blockIdx.y;
    const int m0 = blockIdx.x * MROWS;
    const int Me = cnt[e];
    if (m0 >= Me) return;

    const int lane  = threadIdx.x & 31;
    const int wave  = threadIdx.x >> 5;        // 0..15
    const int strip = wave >> 3;               // 0 or 1: rows m0+strip*16 ..
    const int n0w   = (wave & 7) * 128;        // column group
    const int colB  = lane & 15;
    const int hi    = (lane >> 4) & 1;         // high half-wave = high K range

    __shared__ int   s_ent[MROWS];
    __shared__ float s_red[MROWS][2];
    if (threadIdx.x < MROWS) {
        int pos = m0 + threadIdx.x;
        s_ent[threadIdx.x] = (pos < Me) ? list[e * NTOK + pos] : 0;
        s_red[threadIdx.x][0] = 0.f;
        s_red[threadIdx.x][1] = 0.f;
    }
    __syncthreads();

    const int tokA = s_ent[strip * 16 + (lane & 15)] >> 1;  // gathered token row
    const unsigned short* aptr  = xb  + (size_t)tokA * DIM;
    const unsigned short* bbase = w1t + ((size_t)e << 20);

    v8f acc[8];
    #pragma unroll
    for (int t = 0; t < 8; ++t)
        #pragma unroll
        for (int r = 0; r < 8; ++r) acc[t][r] = 0.f;

    for (int k0 = 0; k0 < DIM; k0 += 32) {
        FragBF a;   // A 16x32 bf16: lanes<16 K0..15 ; lanes>=16 K16..31
        const unsigned short* ap = aptr + k0 + hi * 16;
        a.q[0] = *reinterpret_cast<const uint4*>(ap);
        a.q[1] = *reinterpret_cast<const uint4*>(ap + 8);
        #pragma unroll
        for (int t = 0; t < 8; ++t) {
            int col = n0w + t * 16 + colB;
            const unsigned short* bp = bbase + (size_t)col * DIM + k0 + hi * 16;
            FragBF b;  // B 32x16 bf16: lanes<16 K0..15 of col ; lanes>=16 K16..31
            b.q[0] = *reinterpret_cast<const uint4*>(bp);
            b.q[1] = *reinterpret_cast<const uint4*>(bp + 8);
            acc[t] = __builtin_amdgcn_wmma_f32_16x16x32_bf16(
                false, a.v, false, b.v, (short)0, acc[t], false, false);
        }
    }

    // bias, then per-row mean / sum-of-squares partials into LDS
    #pragma unroll
    for (int t = 0; t < 8; ++t) {
        float bv = b1[e * DIM + n0w + t * 16 + colB];
        #pragma unroll
        for (int r = 0; r < 8; ++r) acc[t][r] += bv;
    }
    const int rowbase = strip * 16 + (hi ? 8 : 0);
    #pragma unroll
    for (int r = 0; r < 8; ++r) {
        float s = 0.f, q = 0.f;
        #pragma unroll
        for (int t = 0; t < 8; ++t) { float v = acc[t][r]; s += v; q += v * v; }
        atomicAdd(&s_red[rowbase + r][0], s);
        atomicAdd(&s_red[rowbase + r][1], q);
    }
    __syncthreads();

    #pragma unroll
    for (int r = 0; r < 8; ++r) {
        int row = rowbase + r;
        int pos = m0 + row;
        if (pos >= Me) continue;
        float mu  = s_red[row][0] * (1.f / DIM);
        float var = s_red[row][1] * (1.f / DIM) - mu * mu;
        float inv = rsqrtf(var + 1e-5f);
        unsigned short* hb = hbuf + (size_t)s_ent[row] * DIM;
        #pragma unroll
        for (int t = 0; t < 8; ++t) {
            int col = n0w + t * 16 + colB;
            float v = (acc[t][r] - mu) * inv;
            v = fmaf(v, ln_g[e * DIM + col], ln_b[e * DIM + col]);
            v = fmaxf(v, 0.f);
            hb[col] = f32_to_bf16(v);
        }
    }
}

// --------------------------- expert GEMM 2 + scatter -----------------------
__global__ __launch_bounds__(512)
void expert_gemm2_scatter(const unsigned short* __restrict__ hbuf,  // [NPAIR][DIM]
                          const unsigned short* __restrict__ w2t,   // [E][out][in]
                          const float* __restrict__ b2,             // [E][DIM]
                          const float* __restrict__ wpair,          // [NPAIR]
                          const int* __restrict__ cnt, const int* __restrict__ list,
                          float* __restrict__ out) {                // [NTOK][DIM]
    const int e  = blockIdx.y;
    const int m0 = blockIdx.x * MROWS;
    const int Me = cnt[e];
    if (m0 >= Me) return;

    const int lane  = threadIdx.x & 31;
    const int wave  = threadIdx.x >> 5;
    const int strip = wave >> 3;
    const int n0w   = (wave & 7) * 128;
    const int colB  = lane & 15;
    const int hi    = (lane >> 4) & 1;

    __shared__ int s_ent[MROWS];
    if (threadIdx.x < MROWS) {
        int pos = m0 + threadIdx.x;
        s_ent[threadIdx.x] = (pos < Me) ? list[e * NTOK + pos] : 0;
    }
    __syncthreads();

    const int pairA = s_ent[strip * 16 + (lane & 15)];
    const unsigned short* aptr  = hbuf + (size_t)pairA * DIM;
    const unsigned short* bbase = w2t + ((size_t)e << 20);

    v8f acc[8];
    #pragma unroll
    for (int t = 0; t < 8; ++t)
        #pragma unroll
        for (int r = 0; r < 8; ++r) acc[t][r] = 0.f;

    for (int k0 = 0; k0 < DIM; k0 += 32) {
        FragBF a;
        const unsigned short* ap = aptr + k0 + hi * 16;
        a.q[0] = *reinterpret_cast<const uint4*>(ap);
        a.q[1] = *reinterpret_cast<const uint4*>(ap + 8);
        #pragma unroll
        for (int t = 0; t < 8; ++t) {
            int col = n0w + t * 16 + colB;
            const unsigned short* bp = bbase + (size_t)col * DIM + k0 + hi * 16;
            FragBF b;
            b.q[0] = *reinterpret_cast<const uint4*>(bp);
            b.q[1] = *reinterpret_cast<const uint4*>(bp + 8);
            acc[t] = __builtin_amdgcn_wmma_f32_16x16x32_bf16(
                false, a.v, false, b.v, (short)0, acc[t], false, false);
        }
    }

    const int rowbase = strip * 16 + (hi ? 8 : 0);
    #pragma unroll
    for (int r = 0; r < 8; ++r) {
        int pos = m0 + rowbase + r;
        if (pos >= Me) continue;
        int   p = s_ent[rowbase + r];
        int   n = p >> 1;
        float w = wpair[p];
        float* op = out + (size_t)n * DIM;
        #pragma unroll
        for (int t = 0; t < 8; ++t) {
            int col = n0w + t * 16 + colB;
            float v = acc[t][r] + b2[e * DIM + col];
            atomicAdd(&op[col], w * v);   // each token: exactly 2 contributions
        }
    }
}

// ------------------------------- launcher ----------------------------------
extern "C" void kernel_launch(void* const* d_in, const int* in_sizes, int n_in,
                              void* d_out, int out_size, void* d_ws, size_t ws_size,
                              hipStream_t stream) {
    const float* x    = (const float*)d_in[0];
    const float* gw1  = (const float*)d_in[1];
    const float* gb1  = (const float*)d_in[2];
    const float* gw2  = (const float*)d_in[3];
    const float* gb2  = (const float*)d_in[4];
    const float* W1   = (const float*)d_in[5];
    const float* b1   = (const float*)d_in[6];
    const float* ln_g = (const float*)d_in[7];
    const float* ln_b = (const float*)d_in[8];
    const float* W2   = (const float*)d_in[9];
    const float* b2   = (const float*)d_in[10];
    float* out = (float*)d_out;          // [NTOK*DIM] out, then 1 aux-loss slot

    // workspace carve-up (~85 MB total)
    char* w = (char*)d_ws;
    auto alloc = [&](size_t bytes) {
        char* p = w;
        w += (bytes + 255) & ~(size_t)255;
        return p;
    };
    unsigned short* xb    = (unsigned short*)alloc((size_t)NTOK * DIM * 2);
    unsigned short* w1t   = (unsigned short*)alloc((size_t)NEXP * DIM * DIM * 2);
    unsigned short* w2t   = (unsigned short*)alloc((size_t)NEXP * DIM * DIM * 2);
    float*          g     = (float*)alloc((size_t)NTOK * GHID * 4);
    float*          wpair = (float*)alloc((size_t)NPAIR * 4);
    int*            list  = (int*)alloc((size_t)NEXP * NTOK * 4);
    int*            cnt   = (int*)alloc(256);
    unsigned short* hbuf  = (unsigned short*)alloc((size_t)NPAIR * DIM * 2);
    (void)ws_size; (void)n_in; (void)in_sizes;

    // 1) zero output (+aux slot) and routing counters
    zero_f32<<<(out_size + 255) / 256, 256, 0, stream>>>(out, out_size);
    zero_i32<<<1, 256, 0, stream>>>(cnt, NEXP);

    // 2) precision conversion / weight transposes (bf16, column-major weights)
    conv_x_bf16<<<(NTOK * DIM) / 256, 256, 0, stream>>>(x, xb, NTOK * DIM);
    transpose_w_bf16<<<dim3(32, 32, NEXP), dim3(32, 8), 0, stream>>>(W1, w1t);
    transpose_w_bf16<<<dim3(32, 32, NEXP), dim3(32, 8), 0, stream>>>(W2, w2t);

    // 3) gating + routing
    gate_hidden<<<(NTOK * GHID) / 256, 256, 0, stream>>>(x, gw1, gb1, g);
    gate_route<<<(NTOK + 255) / 256, 256, 0, stream>>>(g, gw2, gb2, wpair, list, cnt);

    // 4) expert GEMM1 + LayerNorm + ReLU (routed rows only; tail blocks early-exit)
    expert_gemm1_ln<<<dim3(NTOK / MROWS, NEXP), GWAVES * 32, 0, stream>>>(
        xb, w1t, b1, ln_g, ln_b, cnt, list, hbuf);

    // 5) expert GEMM2 + weighted scatter into out
    expert_gemm2_scatter<<<dim3(NTOK / MROWS, NEXP), GWAVES * 32, 0, stream>>>(
        hbuf, w2t, b2, wpair, cnt, list, out);
}
